// _SelfAttention_23673859736415
// MI455X (gfx1250) — compile-verified, matrix operand order checked by reading, not run on hardware
//
#include <hip/hip_runtime.h>
#include <hip/hip_bf16.h>

typedef __bf16 bf16_t;
typedef __attribute__((ext_vector_type(16))) __bf16 v16bf;
typedef __attribute__((ext_vector_type(8)))  __bf16 v8bf;
typedef __attribute__((ext_vector_type(8)))  float  v8f;

// Async-copy builtin operand types (per hipcc diagnostic: param0 is
// 'vector_size(4*sizeof(int)) int' pointer in the global address space).
typedef int vint4 __attribute__((vector_size(16)));
typedef __attribute__((address_space(1))) vint4 gvint4;  // global int4
typedef __attribute__((address_space(3))) vint4 lvint4;  // LDS int4

#define K_DIM   1024
#define K_SEQ   2048
#define K_BATCH 2
#define K_HEADS 16
#define K_HD    64
#define K_ROWS  (K_BATCH * K_SEQ)   /* 4096 */
#define K_QKVN  (3 * K_DIM)         /* 3072 */

#ifndef __has_builtin
#define __has_builtin(x) 0
#endif

#if __has_builtin(__builtin_amdgcn_global_load_async_to_lds_b128)
#define USE_ASYNC_COPY 1
#else
#define USE_ASYNC_COPY 0
#endif

// ---------------------------------------------------------------------------
// helpers
// ---------------------------------------------------------------------------
__device__ __forceinline__ v8f vzero8() {
  v8f z;
#pragma unroll
  for (int e = 0; e < 8; ++e) z[e] = 0.0f;
  return z;
}

__device__ __forceinline__ v8f wmma_bf16(v16bf a, v16bf b, v8f c) {
  // D = A(16x32 bf16) * B(32x16 bf16) + C(16x16 f32)
  return __builtin_amdgcn_wmma_f32_16x16x32_bf16(
      /*neg_a=*/false, a, /*neg_b=*/false, b,
      /*c_mod=*/(short)0, c, /*reuse_a=*/false, /*reuse_b=*/false);
}

// A/B fragment load for 16-bit 16x32 (resp. 32x16) operand, per ISA layout:
// lane group g (=lane>>4) holds K in [g*8, g*8+8) then [16+g*8, +8).
__device__ __forceinline__ v16bf ldfrag(const bf16_t* p, int g) {
  v8bf lo = *(const v8bf*)(p + g * 8);
  v8bf hi = *(const v8bf*)(p + 16 + g * 8);
  return __builtin_shufflevector(lo, hi, 0, 1, 2, 3, 4, 5, 6, 7,
                                 8, 9, 10, 11, 12, 13, 14, 15);
}

// 32-byte global -> LDS copy: async DMA (ASYNCcnt) when available, else
// VGPR round-trip (ds_store_b128) with identical barrier-visibility rules.
__device__ __forceinline__ void copy32B(bf16_t* ldst, const bf16_t* gsrc) {
#if USE_ASYNC_COPY
  __builtin_amdgcn_global_load_async_to_lds_b128((gvint4*)gsrc, (lvint4*)ldst,
                                                 0, 0);
  __builtin_amdgcn_global_load_async_to_lds_b128((gvint4*)(gsrc + 8),
                                                 (lvint4*)(ldst + 8), 0, 0);
#else
  const uint4* s = (const uint4*)gsrc;
  uint4* d = (uint4*)ldst;
  d[0] = s[0];
  d[1] = s[1];
#endif
}

__device__ __forceinline__ void wait_async_all() {
#if USE_ASYNC_COPY
#if __has_builtin(__builtin_amdgcn_s_wait_asynccnt)
  __builtin_amdgcn_s_wait_asynccnt(0);
#else
  asm volatile("s_wait_asynccnt 0" ::: "memory");
#endif
#endif
}

// ---------------------------------------------------------------------------
// 1) LayerNorm (f32 in) -> bf16 out
// ---------------------------------------------------------------------------
__global__ void __launch_bounds__(256)
ln_bf16_kernel(const float* __restrict__ x, const float* __restrict__ gamma,
               const float* __restrict__ beta, bf16_t* __restrict__ out) {
  const int row = blockIdx.x;
  const int t = threadIdx.x;
  const float4 v = ((const float4*)(x + (size_t)row * K_DIM))[t];
  float s = v.x + v.y + v.z + v.w;
  float sq = v.x * v.x + v.y * v.y + v.z * v.z + v.w * v.w;
#pragma unroll
  for (int off = 1; off < 32; off <<= 1) {
    s += __shfl_xor(s, off, 32);
    sq += __shfl_xor(sq, off, 32);
  }
  __shared__ float red_s[8], red_q[8];
  const int w = t >> 5, lane = t & 31;
  if (lane == 0) { red_s[w] = s; red_q[w] = sq; }
  __syncthreads();
  s = 0.0f; sq = 0.0f;
#pragma unroll
  for (int i = 0; i < 8; ++i) { s += red_s[i]; sq += red_q[i]; }
  const float mean = s * (1.0f / K_DIM);
  const float var = sq * (1.0f / K_DIM) - mean * mean;
  const float rstd = rsqrtf(var + 1e-5f);
  const float4 g4 = ((const float4*)gamma)[t];
  const float4 b4 = ((const float4*)beta)[t];
  bf16_t* o = out + (size_t)row * K_DIM + t * 4;
  o[0] = (bf16_t)((v.x - mean) * rstd * g4.x + b4.x);
  o[1] = (bf16_t)((v.y - mean) * rstd * g4.y + b4.y);
  o[2] = (bf16_t)((v.z - mean) * rstd * g4.z + b4.z);
  o[3] = (bf16_t)((v.w - mean) * rstd * g4.w + b4.w);
}

// ---------------------------------------------------------------------------
// 2) f32 -> bf16 weight conversion
// ---------------------------------------------------------------------------
__global__ void __launch_bounds__(256)
f32_to_bf16_kernel(const float* __restrict__ src, bf16_t* __restrict__ dst,
                   int n) {
  for (int i = blockIdx.x * blockDim.x + threadIdx.x; i < n;
       i += gridDim.x * blockDim.x)
    dst[i] = (bf16_t)src[i];
}

// ---------------------------------------------------------------------------
// 128x128 GEMM block tile, double-buffered LDS with async global->LDS copies.
// C[m,n] = sum_k A[m,k] * B[n,k]; A:[M][ldk], B:[N][ldk] bf16 K-contiguous.
// 8 waves in 2x4 grid; each wave computes 64x32 (4x2 WMMA tiles).
// ---------------------------------------------------------------------------
__device__ __forceinline__ void gemm128x128(const bf16_t* __restrict__ A,
                                            const bf16_t* __restrict__ B,
                                            int ldk, int m0, int n0,
                                            v8f acc[4][2]) {
  constexpr int P = 40;  // pitch in bf16 (80B: 16B-aligned, bank-staggered)
  __shared__ bf16_t As[2][128][P];
  __shared__ bf16_t Bs[2][128][P];
  const int tid = threadIdx.x;
  const int lane = tid & 31, w = tid >> 5;
  const int lr = lane & 15, g = lane >> 4;
  const int wm = w & 1, wn = w >> 1;
#pragma unroll
  for (int i = 0; i < 4; ++i)
#pragma unroll
    for (int j = 0; j < 2; ++j) acc[i][j] = vzero8();

  const int row = tid >> 1, hf = tid & 1;  // 2 threads/row, 32B each
  const bf16_t* ga = A + (size_t)(m0 + row) * ldk + hf * 16;
  const bf16_t* gb = B + (size_t)(n0 + row) * ldk + hf * 16;

  // prologue: stage K-slice 0 into buffer 0
  copy32B(&As[0][row][hf * 16], ga);
  copy32B(&Bs[0][row][hf * 16], gb);

  const int NK = ldk >> 5;
  for (int kt = 0; kt < NK; ++kt) {
    const int cur = kt & 1;
    wait_async_all();
    __syncthreads();
    if (kt + 1 < NK) {  // prefetch next K-slice into the other buffer
      copy32B(&As[cur ^ 1][row][hf * 16], ga + (kt + 1) * 32);
      copy32B(&Bs[cur ^ 1][row][hf * 16], gb + (kt + 1) * 32);
    }
    v16bf af[4], bfv[2];
#pragma unroll
    for (int i = 0; i < 4; ++i)
      af[i] = ldfrag(&As[cur][wm * 64 + i * 16 + lr][0], g);
#pragma unroll
    for (int j = 0; j < 2; ++j)
      bfv[j] = ldfrag(&Bs[cur][wn * 32 + j * 16 + lr][0], g);
#pragma unroll
    for (int i = 0; i < 4; ++i)
#pragma unroll
      for (int j = 0; j < 2; ++j)
        acc[i][j] = wmma_bf16(af[i], bfv[j], acc[i][j]);
  }
}

// ---------------------------------------------------------------------------
// 3) QKV GEMM: h[4096,1024] @ w_qkv^T -> scatter bf16 into
//    q[bh][s][d], k[bh][s][d], vT[bh][d][s]
// ---------------------------------------------------------------------------
__global__ void __launch_bounds__(256)
qkv_gemm_kernel(const bf16_t* __restrict__ h, const bf16_t* __restrict__ wq,
                bf16_t* __restrict__ qb, bf16_t* __restrict__ kb,
                bf16_t* __restrict__ vtb) {
  const int m0 = blockIdx.x * 128;
  const int n0 = blockIdx.y * 128;
  v8f acc[4][2];
  gemm128x128(h, wq, K_DIM, m0, n0, acc);

  const int tid = threadIdx.x;
  const int lane = tid & 31, w = tid >> 5;
  const int lr = lane & 15, g = lane >> 4;
  const int wm = w & 1, wn = w >> 1;
#pragma unroll
  for (int i = 0; i < 4; ++i)
#pragma unroll
    for (int j = 0; j < 2; ++j)
#pragma unroll
      for (int r = 0; r < 8; ++r) {
        const int m = m0 + wm * 64 + i * 16 + r + 8 * g;  // C-layout row
        const int n = n0 + wn * 32 + j * 16 + lr;         // C-layout col
        const bf16_t bv = (bf16_t)acc[i][j][r];
        const int part = n >> 10;  // 0=q 1=k 2=v
        const int inner = n & 1023;
        const int hh = inner >> 6, d = inner & 63;
        const int bb = m >> 11, sidx = m & 2047;
        const int bh = bb * K_HEADS + hh;
        if (part == 0)
          qb[((size_t)bh * K_SEQ + sidx) * K_HD + d] = bv;
        else if (part == 1)
          kb[((size_t)bh * K_SEQ + sidx) * K_HD + d] = bv;
        else
          vtb[((size_t)bh * K_HD + d) * K_SEQ + sidx] = bv;
      }
}

// ---------------------------------------------------------------------------
// 4) Flash attention: per (bh, 128 Q rows). Wave w owns rows [16w,16w+16).
//    Double-buffered K / V^T chunks with async global->LDS staging.
// ---------------------------------------------------------------------------
__global__ void __launch_bounds__(256)
attention_kernel(const bf16_t* __restrict__ qbuf, const bf16_t* __restrict__ kbuf,
                 const bf16_t* __restrict__ vtbuf, bf16_t* __restrict__ obuf) {
  constexpr int LP = 72;  // 144B pitch, 16B-aligned
  __shared__ bf16_t Qs[128][LP];
  __shared__ bf16_t Ks[2][64][LP];
  __shared__ bf16_t Vts[2][64][LP];
  __shared__ bf16_t Ps[8][16][LP];

  const int m0 = blockIdx.x * 128;
  const int bh = blockIdx.y;
  const int b = bh >> 4;
  const int hh = bh & 15;
  const int tid = threadIdx.x;
  const int lane = tid & 31, w = tid >> 5;
  const int lr = lane & 15, g = lane >> 4;

  const int qrow = tid >> 1, qhf = tid & 1;   // Q: 2 threads/row, 64B each
  const int krow = tid >> 2, kq4 = tid & 3;   // K/V: 4 threads/row, 32B each
  const bf16_t* gq = qbuf + ((size_t)bh * K_SEQ + m0 + qrow) * K_HD + qhf * 32;
  const bf16_t* gk = kbuf + ((size_t)bh * K_SEQ + krow) * K_HD + kq4 * 16;
  const bf16_t* gv = vtbuf + ((size_t)bh * K_HD + krow) * K_SEQ + kq4 * 16;

  // prologue: stage Q block and KV chunk 0
  copy32B(&Qs[qrow][qhf * 32], gq);
  copy32B(&Qs[qrow][qhf * 32 + 16], gq + 16);
  copy32B(&Ks[0][krow][kq4 * 16], gk);
  copy32B(&Vts[0][krow][kq4 * 16], gv);

  v8f o[4];
  float mrow[8], lrow[8];
#pragma unroll
  for (int t = 0; t < 4; ++t) o[t] = vzero8();
#pragma unroll
  for (int r = 0; r < 8; ++r) { mrow[r] = -1e30f; lrow[r] = 0.0f; }

  for (int jc = 0; jc < K_SEQ; jc += 64) {
    const int ci = (jc >> 6) & 1;
    wait_async_all();
    __syncthreads();
    if (jc + 64 < K_SEQ) {  // prefetch next KV chunk into the other buffer
      copy32B(&Ks[ci ^ 1][krow][kq4 * 16], gk + (size_t)(jc + 64) * K_HD);
      copy32B(&Vts[ci ^ 1][krow][kq4 * 16], gv + (jc + 64));
    }

    // S = Q * K^T over d=64 (2 WMMA k-steps), 4 column tiles
    v8f sfr[4];
#pragma unroll
    for (int t = 0; t < 4; ++t) sfr[t] = vzero8();
#pragma unroll
    for (int ks = 0; ks < 64; ks += 32) {
      v16bf qa = ldfrag(&Qs[w * 16 + lr][ks], g);
#pragma unroll
      for (int t = 0; t < 4; ++t) {
        v16bf kbv = ldfrag(&Ks[ci][t * 16 + lr][ks], g);
        sfr[t] = wmma_bf16(qa, kbv, sfr[t]);
      }
    }
#pragma unroll
    for (int t = 0; t < 4; ++t) sfr[t] *= 0.125f;  // HEAD_DIM^-0.5

    // online softmax: rows live at r + 8*g; reduce across 16 lanes
#pragma unroll
    for (int r = 0; r < 8; ++r) {
      float mx = fmaxf(fmaxf(sfr[0][r], sfr[1][r]), fmaxf(sfr[2][r], sfr[3][r]));
#pragma unroll
      for (int off = 1; off < 16; off <<= 1) mx = fmaxf(mx, __shfl_xor(mx, off, 32));
      const float mnew = fmaxf(mrow[r], mx);
      const float alpha = __expf(mrow[r] - mnew);
      mrow[r] = mnew;
      float ps = 0.0f;
#pragma unroll
      for (int t = 0; t < 4; ++t) {
        const float p = __expf(sfr[t][r] - mnew);
        sfr[t][r] = p;
        ps += p;
      }
#pragma unroll
      for (int off = 1; off < 16; off <<= 1) ps += __shfl_xor(ps, off, 32);
      lrow[r] = lrow[r] * alpha + ps;
#pragma unroll
      for (int t = 0; t < 4; ++t) o[t][r] *= alpha;
    }

    // C-layout P -> A-layout via per-wave LDS scratch (same-wave LDS in order)
#pragma unroll
    for (int t = 0; t < 4; ++t)
#pragma unroll
      for (int r = 0; r < 8; ++r)
        Ps[w][r + 8 * g][t * 16 + lr] = (bf16_t)sfr[t][r];
    __asm__ __volatile__("" ::: "memory");

    // O += P(16x64) * V^T rows d (2 WMMA k-steps over j), 4 d-tiles
#pragma unroll
    for (int ks = 0; ks < 64; ks += 32) {
      v16bf pa = ldfrag(&Ps[w][lr][ks], g);
#pragma unroll
      for (int td = 0; td < 4; ++td) {
        v16bf vb = ldfrag(&Vts[ci][td * 16 + lr][ks], g);
        o[td] = wmma_bf16(pa, vb, o[td]);
      }
    }
  }

  // epilogue: normalize by l and write bf16 [b*SEQ+s][h*64+d]
  float inv[8];
#pragma unroll
  for (int r = 0; r < 8; ++r) inv[r] = 1.0f / lrow[r];
#pragma unroll
  for (int td = 0; td < 4; ++td)
#pragma unroll
    for (int r = 0; r < 8; ++r) {
      const int srow = m0 + w * 16 + r + 8 * g;
      const int col = hh * K_HD + td * 16 + lr;
      obuf[((size_t)b * K_SEQ + srow) * K_DIM + col] = (bf16_t)(o[td][r] * inv[r]);
    }
}

// ---------------------------------------------------------------------------
// 5) Output projection: attn[4096,1024] @ w_out^T + b_out -> f32
// ---------------------------------------------------------------------------
__global__ void __launch_bounds__(256)
out_gemm_kernel(const bf16_t* __restrict__ a, const bf16_t* __restrict__ wo,
                const float* __restrict__ bias, float* __restrict__ out) {
  const int m0 = blockIdx.x * 128;
  const int n0 = blockIdx.y * 128;
  v8f acc[4][2];
  gemm128x128(a, wo, K_DIM, m0, n0, acc);

  const int tid = threadIdx.x;
  const int lane = tid & 31, w = tid >> 5;
  const int lr = lane & 15, g = lane >> 4;
  const int wm = w & 1, wn = w >> 1;
#pragma unroll
  for (int i = 0; i < 4; ++i)
#pragma unroll
    for (int j = 0; j < 2; ++j) {
      const int n = n0 + wn * 32 + j * 16 + lr;
      const float bv = bias[n];
#pragma unroll
      for (int r = 0; r < 8; ++r) {
        const int m = m0 + wm * 64 + i * 16 + r + 8 * g;
        out[(size_t)m * K_DIM + n] = acc[i][j][r] + bv;
      }
    }
}

// ---------------------------------------------------------------------------
// launch
// ---------------------------------------------------------------------------
extern "C" void kernel_launch(void* const* d_in, const int* in_sizes, int n_in,
                              void* d_out, int out_size, void* d_ws,
                              size_t ws_size, hipStream_t stream) {
  (void)in_sizes; (void)n_in; (void)out_size; (void)ws_size;
  const float* x = (const float*)d_in[0];
  const float* gamma = (const float*)d_in[1];
  const float* beta = (const float*)d_in[2];
  const float* wqkv = (const float*)d_in[3];
  const float* wout = (const float*)d_in[4];
  const float* bout = (const float*)d_in[5];
  float* out = (float*)d_out;

  char* ws = (char*)d_ws;
  size_t off = 0;
  auto take = [&](size_t bytes) {
    char* p = ws + off;
    off += (bytes + 255) & ~(size_t)255;
    return p;
  };
  bf16_t* h_bf = (bf16_t*)take((size_t)K_ROWS * K_DIM * 2);
  bf16_t* wq_bf = (bf16_t*)take((size_t)K_QKVN * K_DIM * 2);
  bf16_t* wo_bf = (bf16_t*)take((size_t)K_DIM * K_DIM * 2);
  const size_t head_bytes = (size_t)K_BATCH * K_HEADS * K_SEQ * K_HD * 2;
  bf16_t* q_bf = (bf16_t*)take(head_bytes);
  bf16_t* k_bf = (bf16_t*)take(head_bytes);
  bf16_t* vt_bf = (bf16_t*)take(head_bytes);
  bf16_t* a_bf = (bf16_t*)take((size_t)K_ROWS * K_DIM * 2);

  ln_bf16_kernel<<<dim3(K_ROWS), dim3(256), 0, stream>>>(x, gamma, beta, h_bf);
  f32_to_bf16_kernel<<<dim3(2048), dim3(256), 0, stream>>>(wqkv, wq_bf,
                                                           K_QKVN * K_DIM);
  f32_to_bf16_kernel<<<dim3(1024), dim3(256), 0, stream>>>(wout, wo_bf,
                                                           K_DIM * K_DIM);
  qkv_gemm_kernel<<<dim3(K_ROWS / 128, K_QKVN / 128), dim3(256), 0, stream>>>(
      h_bf, wq_bf, q_bf, k_bf, vt_bf);
  attention_kernel<<<dim3(K_SEQ / 128, K_BATCH * K_HEADS), dim3(256), 0,
                     stream>>>(q_bf, k_bf, vt_bf, a_bf);
  out_gemm_kernel<<<dim3(K_ROWS / 128, K_DIM / 128), dim3(256), 0, stream>>>(
      a_bf, wo_bf, bout, out);
}
